// ChannelsAttentionModule_62723702391454
// MI455X (gfx1250) — matrix operand order repaired and checked by reading, not compiled, not run on hardware
//
#include <hip/hip_runtime.h>

// ---------------------------------------------------------------------------
// ChannelsAttention fused kernel for MI455X (gfx1250, wave32, WMMA).
// One workgroup (256 thr = 8 waves) handles one (b, f) pair:
//   K/Q/V projections -> P = complex K^T Q -> column softmax -> O = V^T W
// All intermediates live in LDS; HBM traffic is just x in + out (~268MB),
// so the kernel is bandwidth-bound (~11.5us floor at 23.3 TB/s).
// Matmuls use v_wmma_f32_16x16x32_f16 (f32 accumulate).
// LDS = 163,584 B <= 320KB/2  ->  2 workgroups resident per WGP.
// ---------------------------------------------------------------------------

typedef __attribute__((ext_vector_type(16))) _Float16 v16h;
typedef __attribute__((ext_vector_type(8)))  _Float16 v8h;
typedef __attribute__((ext_vector_type(8)))  float    v8f;

#define B_  8
#define C_  128
#define F_  256
#define T_  128
#define D_  32

// LDS row strides (halves), padded to break bank conflicts, 16B-aligned rows.
#define XS  136   // X / W rows of 128
#define KQS 40    // K/Q rows of 32
#define VTS 72    // V^T rows of 64
#define SWS 72    // softmax-weight rows of 64
#define PS  65    // P rows of 64 floats, stride 65 -> conflict-free columns

#define SMEM_BYTES 163584

__device__ __forceinline__ v16h ld_a(const _Float16* tile, int stride, int lane) {
  // A operand 16x32 f16: lane holds row m=lane%16; k-chunks of 8 at
  // k0=(lane/16)*8 (halves 0..7) and k0+16 (halves 8..15).
  const _Float16* p = tile + (lane & 15) * stride + ((lane >> 4) << 3);
  union { v16h v; v8h h[2]; } u;
  u.h[0] = *reinterpret_cast<const v8h*>(p);
  u.h[1] = *reinterpret_cast<const v8h*>(p + 16);
  return u.v;
}

__device__ __forceinline__ v16h ld_b(const _Float16* tile, int stride, int lane) {
  // B operand 32x16 f16 from B^T row-major storage: lane holds column
  // n=lane%16; k-chunks of 8 at k0=(lane/16)*8 (halves 0..7) and k0+16 (halves 8..15).
  const _Float16* p = tile + (lane & 15) * stride + ((lane >> 4) << 3);
  union { v16h v; v8h h[2]; } u;
  u.h[0] = *reinterpret_cast<const v8h*>(p);
  u.h[1] = *reinterpret_cast<const v8h*>(p + 16);
  return u.v;
}

__device__ __forceinline__ v8f wmma16(v16h a, v16h b, v8f c) {
  return __builtin_amdgcn_wmma_f32_16x16x32_f16(false, a, false, b, (short)0, c,
                                                false, false);
}

__device__ __forceinline__ void cvt4(_Float16* d, float4 v) {
  d[0] = (_Float16)v.x; d[1] = (_Float16)v.y;
  d[2] = (_Float16)v.z; d[3] = (_Float16)v.w;
}

__global__ void __launch_bounds__(256)
cam_fused_kernel(const float* __restrict__ x,
                 const float* __restrict__ Wk, const float* __restrict__ bk,
                 const float* __restrict__ Wq, const float* __restrict__ bq,
                 const float* __restrict__ Wv, const float* __restrict__ bv,
                 float* __restrict__ out) {
  const int f = blockIdx.x;            // 0..255
  const int b = blockIdx.y;            // 0..7
  const int tid = threadIdx.x;
  const int lane = tid & 31;
  // Wave id: force scalar so tile selection math/branches stay on SALU.
  const int w = __builtin_amdgcn_readfirstlane(tid) >> 5;   // 0..7

  extern __shared__ char smem[];
  _Float16* sXr  = reinterpret_cast<_Float16*>(smem);       // 64 x XS
  _Float16* sXi  = sXr  + 64 * XS;                          // 64 x XS
  _Float16* sWk  = sXi  + 64 * XS;                          // 32 x XS
  _Float16* sWq  = sWk  + 32 * XS;                          // 32 x XS
  _Float16* sWv  = sWq  + 32 * XS;                          // 128 x XS
  _Float16* sKr  = sWv  + 128 * XS;                         // 64 x KQS
  _Float16* sKi  = sKr  + 64 * KQS;
  _Float16* sQr  = sKi  + 64 * KQS;
  _Float16* sQi  = sQr  + 64 * KQS;
  _Float16* sVtr = sQi  + 64 * KQS;                         // 128 x VTS (V^T)
  _Float16* sVti = sVtr + 128 * VTS;
  _Float16* sSWr = sVti + 128 * VTS;                        // 64 x SWS (W^T)
  _Float16* sSWi = sSWr + 64 * SWS;
  float*    sbk  = reinterpret_cast<float*>(sSWi + 64 * SWS);
  float*    sbq  = sbk + 32;
  float*    sbv  = sbq + 32;
  // P scratch ALIASES the X staging area: X is dead after the barrier that
  // precedes stage 3 (its last readers are stages 1/2), P is born in stage 3.
  float*    sP   = reinterpret_cast<float*>(smem);          // 2 x 64 x PS

  // -------- Stage 0: stage weights (transposed layouts fall out naturally:
  // Bt[n][k] = W[d][t] row-major) and X tile into LDS as f16. -------------
  {
    int r  = tid >> 3;                 // 0..31
    int c0 = (tid & 7) * 16;           // 0..112
    const float4* pk = reinterpret_cast<const float4*>(Wk + r * 128 + c0);
    const float4* pq = reinterpret_cast<const float4*>(Wq + r * 128 + c0);
#pragma unroll
    for (int j = 0; j < 4; ++j) {
      cvt4(sWk + r * XS + c0 + j * 4, pk[j]);
      cvt4(sWq + r * XS + c0 + j * 4, pq[j]);
    }
    int rv = tid >> 1;                 // 0..127
    int cv = (tid & 1) * 64;
    const float4* pv = reinterpret_cast<const float4*>(Wv + rv * 128 + cv);
#pragma unroll
    for (int j = 0; j < 16; ++j) cvt4(sWv + rv * XS + cv + j * 4, pv[j]);
    if (tid < 32) { sbk[tid] = bk[tid]; sbq[tid] = bq[tid]; }
    if (tid < 128) sbv[tid] = bv[tid];
  }
  {
    int c  = tid >> 2;                 // 0..63
    int t0 = (tid & 3) * 32;
    const float* xr = x + (((size_t)b * C_ + c) * F_ + f) * T_ + t0;
    const float* xi = xr + (size_t)64 * F_ * T_;
#pragma unroll
    for (int j = 0; j < 8; ++j) {
      float4 vr = reinterpret_cast<const float4*>(xr)[j];
      float4 vi = reinterpret_cast<const float4*>(xi)[j];
      cvt4(sXr + c * XS + t0 + j * 4, vr);
      cvt4(sXi + c * XS + t0 + j * 4, vi);
    }
  }
  __syncthreads();

  // -------- Stage 1: K/Q = X * W^T + bias. 32 tiles (Kr,Ki,Qr,Qi x 4m x 2n)
  for (int it = 0; it < 4; ++it) {
    int tile = w + 8 * it;             // 0..31 (scalar)
    int matsel = tile >> 3;            // 0=Kr 1=Ki 2=Qr 3=Qi
    int sub = tile & 7;
    int mt = sub >> 1, nt = sub & 1;
    const _Float16* Xs = (matsel == 0 || matsel == 2) ? sXr : sXi;
    const _Float16* Ws = (matsel < 2) ? sWk : sWq;
    const float* bias  = (matsel < 2) ? sbk : sbq;
    _Float16* dst = (matsel == 0) ? sKr : (matsel == 1) ? sKi
                  : (matsel == 2) ? sQr : sQi;
    v8f acc = {0.f, 0.f, 0.f, 0.f, 0.f, 0.f, 0.f, 0.f};
#pragma unroll
    for (int ks = 0; ks < 4; ++ks) {
      v16h a  = ld_a(Xs + (mt * 16) * XS + ks * 32, XS, lane);
      v16h bm = ld_b(Ws + (nt * 16) * XS + ks * 32, XS, lane);
      acc = wmma16(a, bm, acc);
    }
    int m0 = mt * 16 + ((lane >> 4) << 3);
    int n  = nt * 16 + (lane & 15);
    float bsc = bias[n];
#pragma unroll
    for (int i = 0; i < 8; ++i)
      dst[(m0 + i) * KQS + n] = (_Float16)(acc[i] + bsc);
  }

  // -------- Stage 2: V = X * Wv^T + bv, stored TRANSPOSED as V^T[t][c]. ---
  for (int it = 0; it < 8; ++it) {
    int tile = w + 8 * it;             // 0..63 (scalar)
    int half = tile >> 5;              // 0=r 1=i
    int sub = tile & 31;
    int mt = sub >> 3, nt = sub & 7;   // mt: c-tiles(4), nt: t-tiles(8)
    const _Float16* Xs = half ? sXi : sXr;
    _Float16* dst = half ? sVti : sVtr;
    v8f acc = {0.f, 0.f, 0.f, 0.f, 0.f, 0.f, 0.f, 0.f};
#pragma unroll
    for (int ks = 0; ks < 4; ++ks) {
      v16h a  = ld_a(Xs + (mt * 16) * XS + ks * 32, XS, lane);
      v16h bm = ld_b(sWv + (nt * 16) * XS + ks * 32, XS, lane);
      acc = wmma16(a, bm, acc);
    }
    int c0 = mt * 16 + ((lane >> 4) << 3);   // 8 contiguous c per lane
    int t  = nt * 16 + (lane & 15);
    float bsc = sbv[t];
    v8h pk;
#pragma unroll
    for (int i = 0; i < 8; ++i) pk[i] = (_Float16)(acc[i] + bsc);
    *reinterpret_cast<v8h*>(dst + t * VTS + c0) = pk;  // one b128 store
  }
  __syncthreads();   // X is dead from here on; sP may reuse its LDS.

  // -------- Stage 3: P_r = Kr Qr^T - Ki Qi^T ; P_i = Kr Qi^T + Ki Qr^T ----
  for (int it = 0; it < 4; ++it) {
    int tile = w + 8 * it;             // 0..31 (scalar)
    int mat = tile >> 4;               // 0=Pr 1=Pi
    int sub = tile & 15;
    int mt = sub >> 2, nt = sub & 3;   // c-tile, e-tile
    const v8f zero = {0.f, 0.f, 0.f, 0.f, 0.f, 0.f, 0.f, 0.f};
    v16h a0 = ld_a(sKr + (mt * 16) * KQS, KQS, lane);
    v16h a1 = ld_a(sKi + (mt * 16) * KQS, KQS, lane);
    v16h b0 = ld_b(((mat == 0) ? sQr : sQi) + (nt * 16) * KQS, KQS, lane);
    v16h b1 = ld_b(((mat == 0) ? sQi : sQr) + (nt * 16) * KQS, KQS, lane);
    v8f acc0 = wmma16(a0, b0, zero);
    v8f acc1 = wmma16(a1, b1, zero);
    float sgn = (mat == 0) ? -1.f : 1.f;   // complex combine, no extra copies
    int m0 = mt * 16 + ((lane >> 4) << 3);
    int n  = nt * 16 + (lane & 15);
    float* Pd = sP + mat * 64 * PS;
#pragma unroll
    for (int i = 0; i < 8; ++i)
      Pd[(m0 + i) * PS + n] = fmaf(sgn, acc1[i], acc0[i]);
  }
  __syncthreads();

  // -------- Stage 4: column softmax over c; write W^T[e][c] as f16. -------
  if (tid < 128) {
    int mat = tid >> 6;                // 0=r 1=i
    int e   = tid & 63;
    float* Pc = sP + mat * 64 * PS + e;
    _Float16* dst = (mat == 0) ? sSWr : sSWi;
    float mx = -3.4e38f;
#pragma unroll 4
    for (int r = 0; r < 64; ++r) mx = fmaxf(mx, Pc[r * PS]);
    float sum = 0.f;
#pragma unroll 4
    for (int r = 0; r < 64; ++r) {
      float ex = __expf(Pc[r * PS] - mx);
      sum += ex;
      Pc[r * PS] = ex;
    }
    float inv = 1.f / sum;
#pragma unroll 4
    for (int r = 0; r < 64; ++r)
      dst[e * SWS + r] = (_Float16)(Pc[r * PS] * inv);
  }
  __syncthreads();

  // -------- Stage 5: O_r = Vr^T Wr - Vi^T Wi ; O_i = Vr^T Wi + Vi^T Wr ----
  for (int it = 0; it < 8; ++it) {
    int tile = w + 8 * it;             // 0..63 (scalar)
    int part = tile >> 5;              // 0=o_r 1=o_i
    int sub = tile & 31;
    int mt = sub >> 2, nt = sub & 3;   // t-tile(8), e-tile(4)
    v8f acc0 = {0.f, 0.f, 0.f, 0.f, 0.f, 0.f, 0.f, 0.f};
    v8f acc1 = {0.f, 0.f, 0.f, 0.f, 0.f, 0.f, 0.f, 0.f};
    const _Float16* Bw0 = (part == 0) ? sSWr : sSWi;
    const _Float16* Bw1 = (part == 0) ? sSWi : sSWr;
#pragma unroll
    for (int ks = 0; ks < 2; ++ks) {
      v16h aR = ld_a(sVtr + (mt * 16) * VTS + ks * 32, VTS, lane);
      v16h aI = ld_a(sVti + (mt * 16) * VTS + ks * 32, VTS, lane);
      v16h b0 = ld_b(Bw0 + (nt * 16) * SWS + ks * 32, SWS, lane);
      v16h b1 = ld_b(Bw1 + (nt * 16) * SWS + ks * 32, SWS, lane);
      acc0 = wmma16(aR, b0, acc0);
      acc1 = wmma16(aI, b1, acc1);
    }
    float sgn = (part == 0) ? -1.f : 1.f;
    int t0 = mt * 16 + ((lane >> 4) << 3);    // 8 contiguous t per lane
    int e  = nt * 16 + (lane & 15);
    size_t obase =
        (((size_t)b * C_ + part * 64 + e) * F_ + f) * T_ + t0;
    float4 o0 = make_float4(fmaf(sgn, acc1[0], acc0[0]),
                            fmaf(sgn, acc1[1], acc0[1]),
                            fmaf(sgn, acc1[2], acc0[2]),
                            fmaf(sgn, acc1[3], acc0[3]));
    float4 o1 = make_float4(fmaf(sgn, acc1[4], acc0[4]),
                            fmaf(sgn, acc1[5], acc0[5]),
                            fmaf(sgn, acc1[6], acc0[6]),
                            fmaf(sgn, acc1[7], acc0[7]));
    *reinterpret_cast<float4*>(out + obase)     = o0;
    *reinterpret_cast<float4*>(out + obase + 4) = o1;
  }
}

extern "C" void kernel_launch(void* const* d_in, const int* in_sizes, int n_in,
                              void* d_out, int out_size, void* d_ws,
                              size_t ws_size, hipStream_t stream) {
  (void)in_sizes; (void)n_in; (void)out_size; (void)d_ws; (void)ws_size;
  const float* x  = (const float*)d_in[0];
  const float* Wk = (const float*)d_in[1];
  const float* bk = (const float*)d_in[2];
  const float* Wq = (const float*)d_in[3];
  const float* bq = (const float*)d_in[4];
  const float* Wv = (const float*)d_in[5];
  const float* bv = (const float*)d_in[6];
  float* out = (float*)d_out;

  // >64KB dynamic LDS: raise the limit (idempotent; not a stream op, so it
  // does not break graph capture).
  static bool attr_set = false;
  if (!attr_set) {
    hipFuncSetAttribute((const void*)cam_fused_kernel,
                        hipFuncAttributeMaxDynamicSharedMemorySize,
                        SMEM_BYTES);
    attr_set = true;
  }

  dim3 grid(F_, B_);   // one WG per (f, b)
  cam_fused_kernel<<<grid, 256, SMEM_BYTES, stream>>>(x, Wk, bk, Wq, bq, Wv,
                                                      bv, out);
}
